// LSTM_Baseline_13219909337285
// MI455X (gfx1250) — compile-verified, matrix operand order checked by reading, not run on hardware
//
#include <hip/hip_runtime.h>

// ---------------------------------------------------------------------------
// MI455X (gfx1250) biLSTM: one wave32 per 16-sequence tile per direction.
// Recurrent + input-side matmuls via v_wmma_f32_16x16x32_f16 (K = H = 32).
// h feedback: D-layout -> LDS (row-major f16) -> A-layout reload (2x b128),
// same chunks streamed to global as the layer-0 activation tensor.
// ---------------------------------------------------------------------------

typedef __attribute__((ext_vector_type(16))) _Float16 v16h;
typedef __attribute__((ext_vector_type(8)))  _Float16 v8h;
typedef __attribute__((ext_vector_type(8)))  float    v8f;

#define SEQ_T   512
#define NSEQ    2048         // B*R
#define HID     32
#define GATES   128          // 4*HID
#define WPB     4            // waves per block (wave32)

__device__ __forceinline__ float fast_sigmoid(float x) {
    // 1/(1+e^-x) = rcp(1 + 2^(-x*log2 e))
    return __builtin_amdgcn_rcpf(1.0f + __builtin_amdgcn_exp2f(-1.4426950408889634f * x));
}
__device__ __forceinline__ float fast_tanh(float x) {
    // tanh(x) = 2*sigmoid(2x) - 1
    return 2.0f * __builtin_amdgcn_rcpf(1.0f + __builtin_amdgcn_exp2f(-2.8853900817779268f * x)) - 1.0f;
}
__device__ __forceinline__ v16h cat16(v8h a, v8h b) {
    v16h r;
#pragma unroll
    for (int i = 0; i < 8; ++i) { r[i] = a[i]; r[i + 8] = b[i]; }
    return r;
}

// ======================= layer 0 (embed dim 1) =============================
__global__ void __launch_bounds__(WPB * 32)
lstm_layer0(const int* __restrict__ ids, const float* __restrict__ embed,
            const float* __restrict__ w_ih, const float* __restrict__ w_hh,
            const float* __restrict__ bias, _Float16* __restrict__ out0)
{
    __shared__ __align__(16) _Float16 hbuf[WPB][16][32];

    const int lane  = threadIdx.x & 31;
    const int wave  = threadIdx.x >> 5;
    const int task  = blockIdx.x * WPB + wave;    // 0..255
    const int dir   = task & 1;
    const int tile  = task >> 1;                  // 0..127
    const int row0  = tile * 16;

    const int n16   = lane & 15;                  // D/B column, A row
    const int kbase = (lane >> 4) * 8;            // K-chunk base (0 or 8)

    // Whh^T in WMMA B layout (8 gate tiles) + per-column input weight/bias.
    v16h  bmat[8];
    float wih_c[8], bias_c[8];
#pragma unroll
    for (int j = 0; j < 8; ++j) {
        const int g = j * 16 + n16;
        wih_c[j]  = w_ih[dir * GATES + g];
        bias_c[j] = bias[dir * GATES + g];
#pragma unroll
        for (int e = 0; e < 16; ++e) {
            const int k = kbase + (e & 7) + (e >> 3) * 16;
            bmat[j][e] = (_Float16)w_hh[(dir * GATES + g) * HID + k];
        }
    }

    v16h ha = {};                  // h_{t-1} in A layout (h0 = 0)
    v8f  cs[2] = {};               // cell state, D layout (cols 0..15 / 16..31)

    // pipelined token-id -> embedding gather (lanes 0..15 own one row each)
    int t0 = dir ? (SEQ_T - 1) : 0;
    float e_cur = 0.0f;
    if (lane < 16) e_cur = embed[ids[(size_t)(row0 + lane) * SEQ_T + t0]];

    for (int s = 0; s < SEQ_T; ++s) {
        const int t = dir ? (SEQ_T - 1 - s) : s;

        // broadcast row embeddings to all lanes (rows r+kbase per D element)
        float er[8];
#pragma unroll
        for (int r = 0; r < 8; ++r)
            er[r] = __shfl(e_cur, r + kbase, 32);

        // prefetch next token id early (independent of the recurrence)
        int id_nxt = 0;
        if (s + 1 < SEQ_T && lane < 16)
            id_nxt = ids[(size_t)(row0 + lane) * SEQ_T + (dir ? t - 1 : t + 1)];

        // gates = (x * Wih + b) + h @ Whh^T   (8 WMMAs, K = 32)
        v8f acc[8];
#pragma unroll
        for (int j = 0; j < 8; ++j) {
            v8f cin;
#pragma unroll
            for (int r = 0; r < 8; ++r)
                cin[r] = fmaf(er[r], wih_c[j], bias_c[j]);
            acc[j] = __builtin_amdgcn_wmma_f32_16x16x32_f16(
                false, ha, false, bmat[j], (short)0, cin, false, false);
        }

        // elementwise (torch gate order i,f,g,o); stage h into LDS as f16
#pragma unroll
        for (int p = 0; p < 2; ++p) {
#pragma unroll
            for (int r = 0; r < 8; ++r) {
                const float ig = fast_sigmoid(acc[0 + p][r]);
                const float fg = fast_sigmoid(acc[2 + p][r]);
                const float gg = fast_tanh   (acc[4 + p][r]);
                const float og = fast_sigmoid(acc[6 + p][r]);
                const float c  = fg * cs[p][r] + ig * gg;
                cs[p][r] = c;
                hbuf[wave][r + kbase][p * 16 + n16] = (_Float16)(og * fast_tanh(c));
            }
        }

        // reload h in A layout (DS in-order within the wave: no barrier)
        const v8h h0 = *(const v8h*)&hbuf[wave][n16][kbase];
        const v8h h1 = *(const v8h*)&hbuf[wave][n16][kbase + 16];
        ha = cat16(h0, h1);

        // stream h_t to global activation tensor [n][t][dir*32 + c] (f16)
        _Float16* orow = out0 + ((size_t)(row0 + n16) * SEQ_T + t) * 64 + dir * HID;
        *(v8h*)(orow + kbase)      = h0;
        *(v8h*)(orow + kbase + 16) = h1;

        // finish gather pipeline for next step
        if (s + 1 < SEQ_T && lane < 16) e_cur = embed[id_nxt];
    }
}

// ======================= layer 1 (input dim 64) ============================
__global__ void __launch_bounds__(WPB * 32)
lstm_layer1(const _Float16* __restrict__ x, const float* __restrict__ w_ih,
            const float* __restrict__ w_hh, const float* __restrict__ bias,
            float* __restrict__ h1last)
{
    __shared__ __align__(16) _Float16 hbuf[WPB][16][32];

    const int lane  = threadIdx.x & 31;
    const int wave  = threadIdx.x >> 5;
    const int task  = blockIdx.x * WPB + wave;
    const int dir   = task & 1;
    const int tile  = task >> 1;
    const int row0  = tile * 16;

    const int n16   = lane & 15;
    const int kbase = (lane >> 4) * 8;

    v16h  bmat[8];        // Whh^T tiles
    v16h  bx[2][8];       // Wih^T tiles, K split 0..31 / 32..63
    float bias_c[8];
#pragma unroll
    for (int j = 0; j < 8; ++j) {
        const int g = j * 16 + n16;
        bias_c[j] = bias[dir * GATES + g];
#pragma unroll
        for (int e = 0; e < 16; ++e) {
            const int k = kbase + (e & 7) + (e >> 3) * 16;
            bmat[j][e]  = (_Float16)w_hh[(dir * GATES + g) * HID + k];
            bx[0][j][e] = (_Float16)w_ih[(dir * GATES + g) * 64 + k];
            bx[1][j][e] = (_Float16)w_ih[(dir * GATES + g) * 64 + 32 + k];
        }
    }

    v16h ha = {};
    v8f  cs[2] = {};
    v8f  hv[2] = {};

    // bwd only needs its t = T-1 output, which is a single reverse-scan step
    const int nsteps = dir ? 1 : SEQ_T;

    // double-buffered A-layout tile of x (4x b128 per lane)
    const int tc = dir ? (SEQ_T - 1) : 0;
    const _Float16* xr = x + ((size_t)(row0 + n16) * SEQ_T + tc) * 64;
    v8h x00 = *(const v8h*)(xr + kbase);
    v8h x01 = *(const v8h*)(xr + kbase + 16);
    v8h x10 = *(const v8h*)(xr + 32 + kbase);
    v8h x11 = *(const v8h*)(xr + 32 + kbase + 16);

    for (int s = 0; s < nsteps; ++s) {
        const v16h xa0 = cat16(x00, x01);
        const v16h xa1 = cat16(x10, x11);

        if (s + 1 < nsteps) {   // prefetch next step's x tile (fwd only)
            const _Float16* xn = x + ((size_t)(row0 + n16) * SEQ_T + (s + 1)) * 64;
            x00 = *(const v8h*)(xn + kbase);
            x01 = *(const v8h*)(xn + kbase + 16);
            x10 = *(const v8h*)(xn + 32 + kbase);
            x11 = *(const v8h*)(xn + 32 + kbase + 16);
        }

        // gates = b + x@Wih^T + h@Whh^T   (24 WMMAs, accumulate in f32)
        v8f acc[8];
#pragma unroll
        for (int j = 0; j < 8; ++j) {
            v8f cin;
#pragma unroll
            for (int r = 0; r < 8; ++r) cin[r] = bias_c[j];
            cin = __builtin_amdgcn_wmma_f32_16x16x32_f16(
                false, xa0, false, bx[0][j], (short)0, cin, false, false);
            cin = __builtin_amdgcn_wmma_f32_16x16x32_f16(
                false, xa1, false, bx[1][j], (short)0, cin, false, false);
            acc[j] = __builtin_amdgcn_wmma_f32_16x16x32_f16(
                false, ha, false, bmat[j], (short)0, cin, false, false);
        }

#pragma unroll
        for (int p = 0; p < 2; ++p) {
#pragma unroll
            for (int r = 0; r < 8; ++r) {
                const float ig = fast_sigmoid(acc[0 + p][r]);
                const float fg = fast_sigmoid(acc[2 + p][r]);
                const float gg = fast_tanh   (acc[4 + p][r]);
                const float og = fast_sigmoid(acc[6 + p][r]);
                const float c  = fg * cs[p][r] + ig * gg;
                cs[p][r] = c;
                const float h  = og * fast_tanh(c);
                hv[p][r] = h;
                hbuf[wave][r + kbase][p * 16 + n16] = (_Float16)h;
            }
        }

        const v8h h0 = *(const v8h*)&hbuf[wave][n16][kbase];
        const v8h h1 = *(const v8h*)&hbuf[wave][n16][kbase + 16];
        ha = cat16(h0, h1);
    }

    // final hidden state (f32) -> h1last[n][dir*32 + col]
#pragma unroll
    for (int p = 0; p < 2; ++p)
#pragma unroll
        for (int r = 0; r < 8; ++r)
            h1last[(size_t)(row0 + r + kbase) * 64 + dir * HID + p * 16 + n16] = hv[p][r];
}

// ============================ head =========================================
// Double softmax over a [B,1] column is identically 1 (faithful), so:
// out[b] = relu(fW0*1 + fW1*mean_r tanh(W . h1last[b*8+r] + bW) + fb)
__global__ void head_kernel(const float* __restrict__ h1last,
                            const float* __restrict__ W, const float* __restrict__ bW,
                            const float* __restrict__ fW, const float* __restrict__ fb,
                            float* __restrict__ out)
{
    const int b = blockIdx.x * blockDim.x + threadIdx.x;
    if (b >= 256) return;
    float sent = 0.0f;
    for (int r = 0; r < 8; ++r) {
        const float* h = h1last + (size_t)(b * 8 + r) * 64;
        float a = bW[0];
#pragma unroll 16
        for (int k = 0; k < 64; ++k) a = fmaf(W[k], h[k], a);
        sent += fast_tanh(a);
    }
    sent *= 0.125f;
    const float v = fmaf(fW[1], sent, fW[0] + fb[0]);
    out[b] = v > 0.0f ? v : 0.0f;
}

// ============================ launcher =====================================
extern "C" void kernel_launch(void* const* d_in, const int* in_sizes, int n_in,
                              void* d_out, int out_size, void* d_ws, size_t ws_size,
                              hipStream_t stream)
{
    (void)in_sizes; (void)n_in; (void)out_size; (void)ws_size;

    const int*   ids   = (const int*)  d_in[5];   // review_ids [B,R,T]
    const float* embed = (const float*)d_in[6];   // [VOCAB,1]
    const float* wih0  = (const float*)d_in[7];   // [2,128,1]
    const float* whh0  = (const float*)d_in[8];   // [2,128,32]
    const float* b0    = (const float*)d_in[9];   // [2,128]
    const float* wih1  = (const float*)d_in[10];  // [2,128,64]
    const float* whh1  = (const float*)d_in[11];  // [2,128,32]
    const float* b1    = (const float*)d_in[12];  // [2,128]
    const float* loW   = (const float*)d_in[13];  // [1,64]
    const float* lob   = (const float*)d_in[14];  // [1]
    const float* fW    = (const float*)d_in[27];  // [1,2]
    const float* fb    = (const float*)d_in[28];  // [1]

    _Float16* out0   = (_Float16*)d_ws;                                   // 128 MB f16 acts
    float*    h1last = (float*)((char*)d_ws + (size_t)NSEQ * SEQ_T * 64 * 2);

    const int blocks = (NSEQ / 16) * 2 / WPB;   // 128 tiles x 2 dirs, 4 waves/block
    lstm_layer0<<<blocks, WPB * 32, 0, stream>>>(ids, embed, wih0, whh0, b0, out0);
    lstm_layer1<<<blocks, WPB * 32, 0, stream>>>(out0, wih1, whh1, b1, h1last);
    head_kernel<<<1, 256, 0, stream>>>(h1last, loW, lob, fW, fb, (float*)d_out);
}
